// LW_PSC_CFFE_86388972192337
// MI455X (gfx1250) — matrix operand order repaired and checked by reading, not compiled, not run on hardware
//
#include <hip/hip_runtime.h>
#include <hip/hip_fp16.h>
#include <math.h>

typedef __attribute__((ext_vector_type(16))) _Float16 v16h;
typedef __attribute__((ext_vector_type(8)))  _Float16 v8h;
typedef __attribute__((ext_vector_type(8)))  float    v8f;

namespace {
constexpr int NB = 2, NC = 20, NX = 256, NY = 256, NZ = 32, CSX = 128, CMID = 49;
constexpr int PAD = 4;                 // halo = max dilation
constexpr int XPD = NX + 2 * PAD;      // 264 (used for both spatial dims)
constexpr int IC1P = 160;              // 148 -> pad 160 (conv1 input = concat(s_x, bev))
constexpr int IC2P = 64;               // 49  -> pad 64
constexpr int ICMP = 160;              // 147 -> pad 160 (wm input = concat(h1,h2,h4))
constexpr int OCMP = 256;              // wm output channels (exact)
constexpr unsigned NEGINF_KEY = 0x007FFFFFu;  // encode(-inf)

// ---- workspace layout (bytes); padded buffers first so one zero pass covers halos ----
constexpr size_t XC_N     = (size_t)NB * XPD * XPD * IC1P;  // f16, padded
constexpr size_t C2_N     = (size_t)NB * XPD * XPD * IC2P;  // f16, padded
constexpr size_t WMIN_N   = (size_t)NB * XPD * XPD * ICMP;  // f16, padded
constexpr size_t WMOUT_N  = (size_t)NB * NX * NY * OCMP;    // f16, unpadded
constexpr size_t BEV_N    = (size_t)NB * NX * NY * NC;      // u32 keys
constexpr size_t XC_OFF   = 0;
constexpr size_t C2_OFF   = XC_OFF  + XC_N  * 2;
constexpr size_t C4_OFF   = C2_OFF  + C2_N  * 2;
constexpr size_t WMIN_OFF = C4_OFF  + C2_N  * 2;
constexpr size_t ZERO_BYTES = WMIN_OFF + WMIN_N * 2;        // zero xc..wmin each call
constexpr size_t WMOUT_OFF= WMIN_OFF + WMIN_N * 2;
constexpr size_t BEV_OFF  = WMOUT_OFF + WMOUT_N * 2;
constexpr size_t WP1_OFF  = BEV_OFF + BEV_N * 4;            // [64][9][160] f16
constexpr size_t WP1_N    = (size_t)64 * 9 * IC1P;
constexpr size_t WP2_OFF  = WP1_OFF + WP1_N * 2;            // [64][9][64]
constexpr size_t WP2_N    = (size_t)64 * 9 * IC2P;
constexpr size_t WP4_OFF  = WP2_OFF + WP2_N * 2;
constexpr size_t WPM_OFF  = WP4_OFF + WP2_N * 2;            // [256][9][160]
constexpr size_t WPM_N    = (size_t)OCMP * 9 * ICMP;
constexpr size_t WPS_OFF  = WPM_OFF + WPM_N * 2;            // [640][128]
constexpr size_t WPS_N    = (size_t)640 * 128;

constexpr size_t SEM_N = (size_t)NB * NC * NY * NX * NZ;
}

__device__ __forceinline__ unsigned enc_key(float f) {
  unsigned u = __float_as_uint(f);
  unsigned m = (u & 0x80000000u) ? 0xFFFFFFFFu : 0x80000000u;
  return u ^ m;
}
__device__ __forceinline__ float dec_key(unsigned k) {
  unsigned m = (k & 0x80000000u) ? 0x80000000u : 0xFFFFFFFFu;
  return __uint_as_float(k ^ m);
}

// A fragment: base pointer already includes +8*g.  idx0..7 = K 8g..8g+7, idx8..15 = K 16+8g..
__device__ __forceinline__ v16h ld_frag_a(const _Float16* p) {
  v8h lo = *(const v8h*)p;
  v8h hi = *(const v8h*)(p + 16);
  return __builtin_shufflevector(lo, hi, 0, 1, 2, 3, 4, 5, 6, 7,
                                 8, 9, 10, 11, 12, 13, 14, 15);
}
// B fragment: base pointer already includes +8*g.  idx0..7 = K 8g..8g+7, idx8..15 = K 16+8g..
__device__ __forceinline__ v16h ld_frag_b(const _Float16* p) {
  v8h lo = *(const v8h*)p;
  v8h hi = *(const v8h*)(p + 16);
  return __builtin_shufflevector(lo, hi, 0, 1, 2, 3, 4, 5, 6, 7,
                                 8, 9, 10, 11, 12, 13, 14, 15);
}
__device__ __forceinline__ v8f wmma_f16(v16h a, v16h b, v8f c) {
  return __builtin_amdgcn_wmma_f32_16x16x32_f16(false, a, false, b, (short)0, c,
                                                false, false);
}

// ---------------- bandwidth kernels ----------------

__global__ void zero16_kernel(uint4* p, size_t n16) {
  size_t i = (size_t)blockIdx.x * blockDim.x + threadIdx.x;
  if (i < n16) p[i] = make_uint4(0u, 0u, 0u, 0u);
}

__global__ void bev_init_kernel(unsigned* bev, size_t n) {
  size_t i = (size_t)blockIdx.x * blockDim.x + threadIdx.x;
  if (i < n) bev[i] = NEGINF_KEY;
}

// One thread per voxel (b,x,y,z): argmax over C, confidence-shifted target,
// scatter-max the full C-vector with order-preserving u32 atomics.
__global__ void scatter_bev_kernel(const float* __restrict__ xp,
                                   const float* __restrict__ cp,
                                   const float* __restrict__ op,
                                   unsigned* __restrict__ bev) {
  size_t t = (size_t)blockIdx.x * blockDim.x + threadIdx.x;
  if (t >= (size_t)NB * NX * NY * NZ) return;
  int z = (int)(t & (NZ - 1));
  size_t q = t >> 5;
  int y = (int)(q & (NY - 1)); q >>= 8;
  int x = (int)(q & (NX - 1));
  int b = (int)(q >> 8);

  float v[NC];
  int best = 0;
  float bv = -3.4e38f;
#pragma unroll
  for (int c = 0; c < NC; ++c) {
    v[c] = xp[(((size_t)(b * NC + c) * NX + x) * NY + y) * NZ + z];
    if (v[c] > bv) { bv = v[c]; best = c; }
  }
  if (best == 0) return;  // semantic == 0 excluded

  size_t pix = ((size_t)b * NX + x) * NY + y;
  // sigmoid(c) > 0.2  <=>  c > ln(0.25)
  float conf = (cp[pix] > -1.3862943611198906f) ? 1.f : 0.f;
  float ox = op[((size_t)b * 2 + 0) * NX * NY + (size_t)x * NY + y];
  float oy = op[((size_t)b * 2 + 1) * NX * NY + (size_t)x * NY + y];
  int gx = (int)floorf((float)x + ox * conf);
  int gy = (int)floorf((float)y + oy * conf);
  if ((unsigned)gx >= (unsigned)NX || (unsigned)gy >= (unsigned)NY) return;

  unsigned* dst = bev + (((size_t)b * NX + gx) * NY + gy) * NC;
#pragma unroll
  for (int c = 0; c < NC; ++c) atomicMax(dst + c, enc_key(v[c]));
}

// xc interior [b][x+4][y+4][ch] = concat(s_x 128ch, decoded bev 20ch, zero pad)
__global__ void build_xc_kernel(const float* __restrict__ sx,
                                const unsigned* __restrict__ bev,
                                _Float16* __restrict__ xc) {
  size_t i = (size_t)blockIdx.x * blockDim.x + threadIdx.x;
  if (i >= (size_t)NB * NX * NY * IC1P) return;
  int ch = (int)(i % IC1P);
  size_t p = i / IC1P;
  int y = (int)(p % NY);
  size_t tq = p / NY;
  int x = (int)(tq % NX);
  int b = (int)(tq / NX);
  float vv = 0.f;
  if (ch < CSX) {
    vv = sx[(((size_t)b * CSX + ch) * NX + x) * NY + y];
  } else if (ch < CSX + NC) {
    unsigned k = bev[p * NC + (ch - CSX)];
    vv = (k <= NEGINF_KEY) ? 0.f : dec_key(k);  // empty cell -> 0 (ref)
  }
  xc[(((size_t)b * XPD + x + PAD) * XPD + y + PAD) * IC1P + ch] = (_Float16)vv;
}

// f32 [OC][ICsrc][taps] -> f16 [OCpad][taps][ICpad], zero padded
__global__ void prepack_w_kernel(const float* __restrict__ w, _Float16* __restrict__ wp,
                                 int oc_n, int ic_n, int icpad, int taps, size_t n) {
  size_t i = (size_t)blockIdx.x * blockDim.x + threadIdx.x;
  if (i >= n) return;
  int ic = (int)(i % icpad);
  int tap = (int)((i / icpad) % taps);
  int oc = (int)(i / ((size_t)icpad * taps));
  float vv = 0.f;
  if (oc < oc_n && ic < ic_n) vv = w[((size_t)oc * ic_n + ic) * taps + tap];
  wp[i] = (_Float16)vv;
}

// ------- WMMA implicit-GEMM 3x3 dilated conv + BN + ReLU, halo-padded input -------
// One wave: 16 output pixels (consecutive y at fixed x) x 64 output channels
// (4 WMMA tiles sharing each A fragment). Loads are unconditional (halo is zero);
// all 4 B fragments are held live so loads batch into one clause per K-chunk and
// the 4 WMMAs issue back-to-back after a single wait.
template <int ICPAD, int DIL>
__global__ __launch_bounds__(256) void conv3x3_wmma_kernel(
    const _Float16* __restrict__ in,   // [B][XPD][XPD][ICPAD]
    const _Float16* __restrict__ wp,   // [ocg_tiles*64][9][ICPAD]
    const float* __restrict__ scale, const float* __restrict__ bias,
    int oc_n, int ocg_tiles,
    _Float16* __restrict__ out1, int o1s, int o1dim, int o1pad, int o1off,
    _Float16* __restrict__ out2, int o2s, int o2dim, int o2pad, int o2off) {
  int wid = blockIdx.x * (blockDim.x >> 5) + (threadIdx.x >> 5);
  int lane = threadIdx.x & 31;
  int ocg = wid % ocg_tiles;
  int t = wid / ocg_tiles;
  int yt = t & 15; t >>= 4;        // NY/16 == 16
  int x = t & (NX - 1);
  int b = t >> 8;
  int g = lane >> 4, lm = lane & 15;
  int y = yt * 16 + lm;            // A-fragment row (pixel) for this lane

  const _Float16* abase =
      in + (((size_t)b * XPD + (x + PAD - DIL)) * XPD + (y + PAD - DIL)) * ICPAD + 8 * g;
  const _Float16* wbase = wp + (size_t)(ocg * 64 + lm) * 9 * ICPAD + 8 * g;
  constexpr size_t WROW = (size_t)16 * 9 * ICPAD;  // 16 output channels of weights

  v8f acc0 = {}, acc1 = {}, acc2 = {}, acc3 = {};
#pragma unroll
  for (int kh = 0; kh < 3; ++kh) {
#pragma unroll
    for (int kw = 0; kw < 3; ++kw) {
      const _Float16* ap = abase + ((size_t)kh * DIL * XPD + kw * DIL) * ICPAD;
      const _Float16* wq = wbase + (size_t)(kh * 3 + kw) * ICPAD;
#pragma unroll
      for (int ic0 = 0; ic0 < ICPAD; ic0 += 32) {
        // batch all loads for this K-chunk, then fire 4 WMMAs back-to-back
        v16h a  = ld_frag_a(ap + ic0);
        v16h b0 = ld_frag_b(wq + ic0);
        v16h b1 = ld_frag_b(wq + WROW + ic0);
        v16h b2 = ld_frag_b(wq + 2 * WROW + ic0);
        v16h b3 = ld_frag_b(wq + 3 * WROW + ic0);
        acc0 = wmma_f16(a, b0, acc0);
        acc1 = wmma_f16(a, b1, acc1);
        acc2 = wmma_f16(a, b2, acc2);
        acc3 = wmma_f16(a, b3, acc3);
      }
    }
  }

  v8f accs[4] = {acc0, acc1, acc2, acc3};
#pragma unroll
  for (int j = 0; j < 4; ++j) {
    int oc = ocg * 64 + j * 16 + lm;
    if (oc >= oc_n) continue;
    float sc = scale[oc], bi = bias[oc];
#pragma unroll
    for (int r = 0; r < 8; ++r) {
      int yy = yt * 16 + r + 8 * g;  // D layout: VGPR r holds M = r + 8*(lane/16)
      float vv = fmaxf(accs[j][r] * sc + bi, 0.f);
      _Float16 hv = (_Float16)vv;
      out1[(((size_t)b * o1dim + x + o1pad) * o1dim + yy + o1pad) * o1s + o1off + oc] = hv;
      if (out2)
        out2[(((size_t)b * o2dim + x + o2pad) * o2dim + yy + o2pad) * o2s + o2off + oc] = hv;
    }
  }
}

// 1x1 conv 128 -> 640 on sem_feat; 4 oc tiles per wave sharing A; permuted f32 store
__global__ __launch_bounds__(256) void head_sem_wmma_kernel(
    const _Float16* __restrict__ feat,   // wm_out [B][X][Y][256], sem = ch 0..127
    const _Float16* __restrict__ wp,     // [640][128] f16
    const float* __restrict__ bsem,
    float* __restrict__ out) {
  int wid = blockIdx.x * (blockDim.x >> 5) + (threadIdx.x >> 5);
  int lane = threadIdx.x & 31;
  int ocg = wid % 10;                    // 640 / 64
  int t = wid / 10;
  int yt = t & 15; t >>= 4;
  int x = t & (NX - 1);
  int b = t >> 8;
  int g = lane >> 4, lm = lane & 15;
  int y = yt * 16 + lm;

  const _Float16* abase = feat + (((size_t)b * NX + x) * NY + y) * OCMP + 8 * g;
  const _Float16* wbase = wp + (size_t)(ocg * 64 + lm) * 128 + 8 * g;
  constexpr size_t WROW = (size_t)16 * 128;

  v8f acc0 = {}, acc1 = {}, acc2 = {}, acc3 = {};
#pragma unroll
  for (int ic0 = 0; ic0 < 128; ic0 += 32) {
    v16h a  = ld_frag_a(abase + ic0);
    v16h b0 = ld_frag_b(wbase + ic0);
    v16h b1 = ld_frag_b(wbase + WROW + ic0);
    v16h b2 = ld_frag_b(wbase + 2 * WROW + ic0);
    v16h b3 = ld_frag_b(wbase + 3 * WROW + ic0);
    acc0 = wmma_f16(a, b0, acc0);
    acc1 = wmma_f16(a, b1, acc1);
    acc2 = wmma_f16(a, b2, acc2);
    acc3 = wmma_f16(a, b3, acc3);
  }

  v8f accs[4] = {acc0, acc1, acc2, acc3};
#pragma unroll
  for (int j = 0; j < 4; ++j) {
    int oc = ocg * 64 + j * 16 + lm;
    float bs = bsem[oc];
    int cc = oc >> 5, z = oc & 31;       // oc = c*Z + z
#pragma unroll
    for (int r = 0; r < 8; ++r) {
      int yy = yt * 16 + r + 8 * g;
      // sem_out[b, c, y, x, z]
      out[((((size_t)b * NC + cc) * NY + yy) * NX + x) * NZ + z] = accs[j][r] + bs;
    }
  }
}

// center (1ch) + offset (2ch) 1x1 heads from ins_feat (wm_out ch 128..255)
__global__ void head_small_kernel(const _Float16* __restrict__ feat,
                                  const float* __restrict__ whm, const float* __restrict__ bhm,
                                  const float* __restrict__ woff, const float* __restrict__ boff,
                                  float* __restrict__ out_c, float* __restrict__ out_o) {
  size_t p = (size_t)blockIdx.x * blockDim.x + threadIdx.x;
  if (p >= (size_t)NB * NX * NY) return;
  const _Float16* f = feat + p * OCMP + 128;
  float a0 = 0.f, a1 = 0.f, a2 = 0.f;
#pragma unroll 4
  for (int ic = 0; ic < 128; ++ic) {
    float vv = (float)f[ic];
    a0 += vv * whm[ic];
    a1 += vv * woff[ic];
    a2 += vv * woff[128 + ic];
  }
  out_c[p] = a0 + bhm[0];
  size_t b = p / ((size_t)NX * NY);
  size_t r = p - b * (size_t)NX * NY;
  out_o[b * 2 * (size_t)NX * NY + r] = a1 + boff[0];
  out_o[b * 2 * (size_t)NX * NY + (size_t)NX * NY + r] = a2 + boff[1];
}

// ---------------- launch ----------------

extern "C" void kernel_launch(void* const* d_in, const int* in_sizes, int n_in,
                              void* d_out, int out_size, void* d_ws, size_t ws_size,
                              hipStream_t stream) {
  (void)in_sizes; (void)n_in; (void)out_size; (void)ws_size;
  const float* x_pred = (const float*)d_in[0];
  const float* s_x    = (const float*)d_in[1];
  const float* center = (const float*)d_in[2];
  const float* offset = (const float*)d_in[3];
  const float* w1 = (const float*)d_in[4];
  const float* s1 = (const float*)d_in[5];  const float* b1 = (const float*)d_in[6];
  const float* w2 = (const float*)d_in[7];
  const float* s2 = (const float*)d_in[8];  const float* b2 = (const float*)d_in[9];
  const float* w4 = (const float*)d_in[10];
  const float* s4 = (const float*)d_in[11]; const float* b4 = (const float*)d_in[12];
  const float* wm = (const float*)d_in[13];
  const float* sm = (const float*)d_in[14]; const float* bm = (const float*)d_in[15];
  const float* w_sem = (const float*)d_in[16]; const float* b_sem = (const float*)d_in[17];
  const float* w_hm  = (const float*)d_in[18]; const float* b_hm  = (const float*)d_in[19];
  const float* w_off = (const float*)d_in[20]; const float* b_off = (const float*)d_in[21];

  char* ws = (char*)d_ws;
  _Float16* xc     = (_Float16*)(ws + XC_OFF);
  _Float16* c2in   = (_Float16*)(ws + C2_OFF);
  _Float16* c4in   = (_Float16*)(ws + C4_OFF);
  _Float16* wmin   = (_Float16*)(ws + WMIN_OFF);
  _Float16* wmout  = (_Float16*)(ws + WMOUT_OFF);
  unsigned* bev    = (unsigned*)(ws + BEV_OFF);
  _Float16* wp1    = (_Float16*)(ws + WP1_OFF);
  _Float16* wp2    = (_Float16*)(ws + WP2_OFF);
  _Float16* wp4    = (_Float16*)(ws + WP4_OFF);
  _Float16* wpm    = (_Float16*)(ws + WPM_OFF);
  _Float16* wps    = (_Float16*)(ws + WPS_OFF);
  float* out = (float*)d_out;
  float* out_center = out + SEM_N;
  float* out_offset = out_center + (size_t)NB * NX * NY;

  const int T = 256;

  // zero all padded feature buffers (halos + padding channels)
  {
    size_t n16 = ZERO_BYTES / 16;
    zero16_kernel<<<(unsigned)((n16 + T - 1) / T), T, 0, stream>>>((uint4*)ws, n16);
  }
  bev_init_kernel<<<(unsigned)((BEV_N + T - 1) / T), T, 0, stream>>>(bev, BEV_N);

  scatter_bev_kernel<<<(unsigned)(((size_t)NB * NX * NY * NZ) / T), T, 0, stream>>>(
      x_pred, center, offset, bev);

  build_xc_kernel<<<(unsigned)(((size_t)NB * NX * NY * IC1P + T - 1) / T), T, 0, stream>>>(
      s_x, bev, xc);

  prepack_w_kernel<<<(unsigned)((WP1_N + T - 1) / T), T, 0, stream>>>(w1, wp1, CMID, CSX + NC, IC1P, 9, WP1_N);
  prepack_w_kernel<<<(unsigned)((WP2_N + T - 1) / T), T, 0, stream>>>(w2, wp2, CMID, CMID, IC2P, 9, WP2_N);
  prepack_w_kernel<<<(unsigned)((WP2_N + T - 1) / T), T, 0, stream>>>(w4, wp4, CMID, CMID, IC2P, 9, WP2_N);
  prepack_w_kernel<<<(unsigned)((WPM_N + T - 1) / T), T, 0, stream>>>(wm, wpm, OCMP, 3 * CMID, ICMP, 9, WPM_N);
  prepack_w_kernel<<<(unsigned)((WPS_N + T - 1) / T), T, 0, stream>>>(w_sem, wps, 640, CSX, 128, 1, WPS_N);

  // conv1: xc(160) -> h1(49); feeds conv2 input and wm_in slice [0..48]
  {
    unsigned blocks = (unsigned)((size_t)NB * NX * 16 * 1 / 8);
    conv3x3_wmma_kernel<IC1P, 1><<<blocks, 256, 0, stream>>>(
        xc, wp1, s1, b1, CMID, 1,
        c2in, IC2P, XPD, PAD, 0, wmin, ICMP, XPD, PAD, 0);
  }
  // conv2 (dil 2): c2in(64) -> h2(49); feeds conv4 input and wm_in slice [49..97]
  {
    unsigned blocks = (unsigned)((size_t)NB * NX * 16 * 1 / 8);
    conv3x3_wmma_kernel<IC2P, 2><<<blocks, 256, 0, stream>>>(
        c2in, wp2, s2, b2, CMID, 1,
        c4in, IC2P, XPD, PAD, 0, wmin, ICMP, XPD, PAD, CMID);
  }
  // conv4 (dil 4): c4in(64) -> h4(49); wm_in slice [98..146]
  {
    unsigned blocks = (unsigned)((size_t)NB * NX * 16 * 1 / 8);
    conv3x3_wmma_kernel<IC2P, 4><<<blocks, 256, 0, stream>>>(
        c4in, wp4, s4, b4, CMID, 1,
        wmin, ICMP, XPD, PAD, 2 * CMID, (_Float16*)nullptr, 0, 0, 0, 0);
  }
  // wm: wmin(160) -> wmout(256), 4 oc groups of 64
  {
    unsigned blocks = (unsigned)((size_t)NB * NX * 16 * 4 / 8);
    conv3x3_wmma_kernel<ICMP, 1><<<blocks, 256, 0, stream>>>(
        wmin, wpm, sm, bm, OCMP, 4,
        wmout, OCMP, NX, 0, 0, (_Float16*)nullptr, 0, 0, 0, 0);
  }
  // sem head: 1x1 GEMM 128 -> 640, permuted f32 store
  {
    unsigned blocks = (unsigned)((size_t)NB * NX * 16 * 10 / 8);
    head_sem_wmma_kernel<<<blocks, 256, 0, stream>>>(wmout, wps, b_sem, out);
  }
  // center + offset heads
  head_small_kernel<<<(unsigned)(((size_t)NB * NX * NY + T - 1) / T), T, 0, stream>>>(
      wmout, w_hm, b_hm, w_off, b_off, out_center, out_offset);
}
